// LSTM_9852654977358
// MI455X (gfx1250) — matrix operand order repaired and checked by reading, not compiled, not run on hardware
//
#include <hip/hip_runtime.h>
#include <hip/hip_bf16.h>
#include <stdint.h>

// ---------------------------------------------------------------------------
// LSTM (B=256, T=4096, I=41, H=20) + per-timestep MLP head, for gfx1250.
//
//   K1: xg[b,t,0:80] = x @ W_ih^T + (b_ih+b_hh)      f32 WMMA 16x16x4,
//       B operand + bias held in registers, 8 M-tiles per wave.
//   K2: sequential scan over T, 16 sequences/wave     f16 WMMA 16x16x32 recurrent,
//       TDM double-buffered xg prefetch, contiguous per-lane elementwise.
//   K3: MLP head tanh/gelu/tanh/sigmoid               parallel VALU.
//
// Workspace: xg = BT*80*4 = 335.5 MB, r_out = BT*20*4 = 83.9 MB  (~420 MB of d_ws)
// ---------------------------------------------------------------------------

typedef __attribute__((ext_vector_type(2)))  float    v2f;
typedef __attribute__((ext_vector_type(8)))  float    v8f;
typedef __attribute__((ext_vector_type(16))) _Float16 v16h;
typedef __attribute__((ext_vector_type(8)))  _Float16 v8h;
typedef __attribute__((ext_vector_type(4)))  unsigned v4u;
typedef __attribute__((ext_vector_type(8)))  int      v8i;
typedef __attribute__((ext_vector_type(4)))  int      v4i;

#define BB   256
#define TT   4096
#define II   41
#define HH   20
#define GG   80           // 4*H
#define BT   (256LL*4096LL)

// ---- fast transcendentals (v_exp_f32 / v_rcp_f32) -------------------------
__device__ __forceinline__ float fexp2(float x) { return __builtin_amdgcn_exp2f(x); }
__device__ __forceinline__ float frcp(float x)  { return __builtin_amdgcn_rcpf(x); }
__device__ __forceinline__ float fsig(float x) {
    return frcp(1.0f + fexp2(-1.4426950408889634f * x));
}
__device__ __forceinline__ float ftanh(float x) {
    // tanh(x) = 2*sigmoid(2x) - 1
    return 2.0f * frcp(1.0f + fexp2(-2.8853900817779268f * x)) - 1.0f;
}

// ===========================================================================
// K1: xg = x @ W_ih^T + (b_ih + b_hh).  M = B*T, K = 41 (pad 44), N = 80.
// One wave processes 8 consecutive 16-row M tiles with B (W_ih^T) and bias
// kept entirely in registers: 55 v2f B fragments + 5 bias scalars.
// ===========================================================================
#define K1_TPW 8   // M tiles per wave

__global__ void __launch_bounds__(256)
k1_xg_gemm(const float* __restrict__ x, const float* __restrict__ W_ih,
           const float* __restrict__ b_ih, const float* __restrict__ b_hh,
           float* __restrict__ xg)
{
    const int lane = threadIdx.x & 31;
    const int wave = threadIdx.x >> 5;
    const long long w = (long long)blockIdx.x * 8 + wave;   // 8192 waves
    const int row  = lane & 15;
    const int khi  = (lane >= 16) ? 2 : 0;   // f32 A/B layout: lanes 16-31 hold K+2
    const int mrow = (lane >= 16) ? 8 : 0;   // C/D layout

    // ---- preload B operand fragments and bias into registers ----
    v2f   breg[11][5];
    float bias[5];
#pragma unroll
    for (int nt = 0; nt < 5; ++nt) {
        const int n = nt * 16 + row;
        bias[nt] = b_ih[n] + b_hh[n];
#pragma unroll
        for (int kk = 0; kk < 11; ++kk) {
            const int k0 = kk * 4 + khi;
            v2f b;
            b.x = (k0     < II) ? W_ih[n * II + k0]     : 0.0f;
            b.y = (k0 + 1 < II) ? W_ih[n * II + k0 + 1] : 0.0f;
            breg[kk][nt] = b;
        }
    }

    for (int it = 0; it < K1_TPW; ++it) {
        const long long m0 = (w * K1_TPW + it) * 16;
        const float* xrow = x + (m0 + row) * II;

        v8f acc[5] = {};
#pragma unroll
        for (int kk = 0; kk < 11; ++kk) {
            const int k0 = kk * 4 + khi;
            v2f a;
            a.x = (k0     < II) ? xrow[k0]     : 0.0f;
            a.y = (k0 + 1 < II) ? xrow[k0 + 1] : 0.0f;
#pragma unroll
            for (int nt = 0; nt < 5; ++nt)
                acc[nt] = __builtin_amdgcn_wmma_f32_16x16x4_f32(
                    false, a, false, breg[kk][nt], (short)0, acc[nt], false, false);
        }

#pragma unroll
        for (int nt = 0; nt < 5; ++nt) {
            const int n = nt * 16 + row;
#pragma unroll
            for (int j = 0; j < 8; ++j)
                xg[(m0 + mrow + j) * GG + n] = acc[nt][j] + bias[nt];
        }
    }
}

// ===========================================================================
// K2: sequential LSTM scan.  One wave (32 lanes) per 16 batch sequences.
// Per step: 5x V_WMMA_F32_16X16X32_F16 (h@W_hh^T), gates via LDS, h to LDS+global.
// xg tiles (16x80 f32, rows strided T*80) prefetched by TDM, double buffered.
//
// Elementwise mapping: lane -> (row = lane&15, cols = (lane>=16 ? 10:0)..+9)
// so every lane touches CONTIGUOUS LDS/global ranges (vectorizable DS ops).
//
// Dynamic LDS layout (16384 B):
//   [0      .. 10240)  xg double buffer: 2 x 1280 f32
//   [10240  .. 15360)  rec[16][80] f32  (recurrent WMMA result)
//   [15360  .. 16384)  hbuf[16][32] f16 (h, K padded to 32 with zeros)
// ===========================================================================
#define XG_OFF   0u
#define XG_BYTES 5120u
#define REC_OFF  10240u
#define H_OFF    15360u

extern __shared__ char k2_smem[];

__device__ __forceinline__ void tdm_issue_xg(const float* xg_base, long long b0,
                                             int t, const v8i g1)
{
    const unsigned long long ga = (unsigned long long)(uintptr_t)xg_base
        + ((unsigned long long)b0 * TT + (unsigned)t) * (unsigned long long)(GG * 4);
    const unsigned lds = XG_OFF + ((unsigned)t & 1u) * XG_BYTES;
    v4u g0;
    g0.x = 1u;                                           // count=1, user mode
    g0.y = lds;                                          // lds_addr (bytes)
    g0.z = (unsigned)(ga & 0xFFFFFFFFull);               // global_addr low
    g0.w = (unsigned)((ga >> 32) & 0x1FFFFFFull) | (2u << 30);  // addr hi + type=2
    v4i zz4 = {0, 0, 0, 0};
    v8i zz8 = {0, 0, 0, 0, 0, 0, 0, 0};
    __builtin_amdgcn_tensor_load_to_lds(g0, g1, zz4, zz4, zz8, 0);
}

__global__ void __launch_bounds__(32)
k2_scan(const float* __restrict__ xg, const float* __restrict__ W_hh,
        float* __restrict__ rout)
{
    float*    xgb = (float*)(k2_smem + XG_OFF);
    float*    rec = (float*)(k2_smem + REC_OFF);
    _Float16* hb  = (_Float16*)(k2_smem + H_OFF);

    const int lane = threadIdx.x;                      // 32 threads, one wave
    const long long b0 = (long long)blockIdx.x * 16;   // 16 blocks cover B=256

    // zero h buffer (cols 20..31 stay zero forever -> K padding)
    for (int i = lane; i < 16 * 32; i += 32) hb[i] = (_Float16)0.0f;

    // preload B operand (W_hh^T as 32x16-per-tile f16) into registers: 5 x v16h
    const int nloc  = lane & 15;
    const int kbase = (lane >= 16) ? 16 : 0;
    v16h breg[5];
#pragma unroll
    for (int nt = 0; nt < 5; ++nt) {
        const int n = nt * 16 + nloc;
#pragma unroll
        for (int i = 0; i < 16; ++i) {
            const int k = (i < 8) ? (kbase >> 1) + i : 16 + (kbase >> 1) + (i - 8);
            breg[nt][i] = (_Float16)((k < HH) ? W_hh[n * HH + k] : 0.0f);
        }
    }

    // loop-invariant TDM descriptor group 1:
    //   data_size=4B, tensor_dim0=80, tensor_dim1=16, tile=80x16, stride0=T*80
    v8i g1;
    g1[0] = (int)(2u << 16);
    g1[1] = (int)((unsigned)GG << 16);
    g1[2] = (int)(16u << 16);
    g1[3] = (int)((unsigned)GG << 16);
    g1[4] = 16;
    g1[5] = GG * TT;       // 327680 elements
    g1[6] = 0;
    g1[7] = 0;

    float creg[10];
#pragma unroll
    for (int j = 0; j < 10; ++j) creg[j] = 0.0f;

    // prefetch xg tile for t=0
    tdm_issue_xg(xg, b0, 0, g1);

    const int arow = lane & 15;
    const int koff = (lane >= 16) ? 8 : 0;
    const int mrow = (lane >= 16) ? 8 : 0;
    const int r    = lane & 15;                 // elementwise: my row
    const int c0   = (lane >= 16) ? 10 : 0;     // elementwise: my column slice

    for (int t = 0; t < TT; ++t) {
        if (t + 1 < TT) tdm_issue_xg(xg, b0, t + 1, g1);   // uniform branch

        // A operand: h (f16, A layout) from LDS -- 2x 16B ds loads per lane
        v8h alo = *(const v8h*)(hb + arow * 32 + koff);
        v8h ahi = *(const v8h*)(hb + arow * 32 + 16 + koff);
        v16h a;
#pragma unroll
        for (int i = 0; i < 8; ++i) { a[i] = alo[i]; a[8 + i] = ahi[i]; }

        // recurrent gates: 5 independent WMMAs, C = 0
        v8f acc[5];
#pragma unroll
        for (int nt = 0; nt < 5; ++nt) {
            v8f z = {};
            acc[nt] = __builtin_amdgcn_wmma_f32_16x16x32_f16(
                false, a, false, breg[nt], (short)0, z, false, false);
        }

        // scatter recurrent result to LDS (C/D layout -> row-major rec[16][80])
#pragma unroll
        for (int nt = 0; nt < 5; ++nt) {
            const int n = nt * 16 + nloc;
#pragma unroll
            for (int j = 0; j < 8; ++j)
                rec[(mrow + j) * GG + n] = acc[nt][j];
        }

        // wait for this step's xg tile (one more TDM op may be in flight)
        if (t + 1 < TT) __builtin_amdgcn_s_wait_tensorcnt(1);
        else            __builtin_amdgcn_s_wait_tensorcnt(0);

        const float* xgt = xgb + ((unsigned)t & 1u) * (XG_BYTES / 4);
        const float* xr  = xgt + r * GG;
        const float* rr  = rec + r * GG;
        _Float16*    hr  = hb  + r * 32;
        float*       ror = rout + ((b0 + r) * TT + t) * HH;

        // elementwise LSTM cell: 10 contiguous columns per lane
#pragma unroll
        for (int j = 0; j < 10; ++j) {
            const int cc = c0 + j;
            const float iv = fsig (xr[cc     ] + rr[cc     ]);
            const float fv = fsig (xr[cc + 20] + rr[cc + 20]);
            const float gv = ftanh(xr[cc + 40] + rr[cc + 40]);
            const float ov = fsig (xr[cc + 60] + rr[cc + 60]);
            const float c  = fv * creg[j] + iv * gv;
            creg[j] = c;
            const float h = ov * ftanh(c);
            hr[cc]  = (_Float16)h;     // feeds next step's A operand
            ror[cc] = h;               // r_out (contiguous 40B per lane)
        }
        // single wave: DS ops are in-order, no barrier needed before next A load
    }
}

// ===========================================================================
// K3: MLP head per (b,t): 20 -> tanh(10) -> gelu(10) -> tanh(10) -> sigmoid(1)
// ===========================================================================
__global__ void __launch_bounds__(256)
k3_mlp(const float* __restrict__ rout,
       const float* __restrict__ Wh, const float* __restrict__ bh,
       const float* __restrict__ W1, const float* __restrict__ b1,
       const float* __restrict__ W2, const float* __restrict__ b2,
       const float* __restrict__ W3, const float* __restrict__ b3,
       float* __restrict__ out, long long n)
{
    __shared__ float s[441];
    for (int i = threadIdx.x; i < 200; i += blockDim.x) s[i]       = Wh[i];
    for (int i = threadIdx.x; i < 10;  i += blockDim.x) s[200 + i] = bh[i];
    for (int i = threadIdx.x; i < 100; i += blockDim.x) s[210 + i] = W1[i];
    for (int i = threadIdx.x; i < 10;  i += blockDim.x) s[310 + i] = b1[i];
    for (int i = threadIdx.x; i < 100; i += blockDim.x) s[320 + i] = W2[i];
    for (int i = threadIdx.x; i < 10;  i += blockDim.x) s[420 + i] = b2[i];
    for (int i = threadIdx.x; i < 10;  i += blockDim.x) s[430 + i] = W3[i];
    if (threadIdx.x == 0) s[440] = b3[0];
    __syncthreads();

    const long long idx = (long long)blockIdx.x * blockDim.x + threadIdx.x;
    if (idx >= n) return;

    float h[HH];
#pragma unroll
    for (int i = 0; i < HH; ++i) h[i] = rout[idx * HH + i];

    float z1[10];
#pragma unroll
    for (int o = 0; o < 10; ++o) {
        float acc = s[200 + o];
#pragma unroll
        for (int i = 0; i < HH; ++i) acc += s[o * HH + i] * h[i];
        z1[o] = ftanh(acc);
    }
    float z2[10];
#pragma unroll
    for (int o = 0; o < 10; ++o) {
        float acc = s[310 + o];
#pragma unroll
        for (int i = 0; i < 10; ++i) acc += s[210 + o * 10 + i] * z1[i];
        z2[o] = 0.5f * acc * (1.0f + erff(acc * 0.7071067811865476f));  // exact gelu
    }
    float z3[10];
#pragma unroll
    for (int o = 0; o < 10; ++o) {
        float acc = s[420 + o];
#pragma unroll
        for (int i = 0; i < 10; ++i) acc += s[320 + o * 10 + i] * z2[i];
        z3[o] = ftanh(acc);
    }
    float acc = s[440];
#pragma unroll
    for (int i = 0; i < 10; ++i) acc += s[430 + i] * z3[i];
    out[idx] = fsig(acc);
}

// ===========================================================================
extern "C" void kernel_launch(void* const* d_in, const int* in_sizes, int n_in,
                              void* d_out, int out_size, void* d_ws, size_t ws_size,
                              hipStream_t stream)
{
    const float* x    = (const float*)d_in[0];
    const float* W_ih = (const float*)d_in[1];
    const float* W_hh = (const float*)d_in[2];
    const float* b_ih = (const float*)d_in[3];
    const float* b_hh = (const float*)d_in[4];
    const float* Wh   = (const float*)d_in[5];
    const float* bh   = (const float*)d_in[6];
    const float* W1   = (const float*)d_in[7];
    const float* b1   = (const float*)d_in[8];
    const float* W2   = (const float*)d_in[9];
    const float* b2   = (const float*)d_in[10];
    const float* W3   = (const float*)d_in[11];
    const float* b3   = (const float*)d_in[12];
    float* out = (float*)d_out;

    float* xg   = (float*)d_ws;                                  // BT*80 f32
    float* rout = (float*)((char*)d_ws + (size_t)BT * GG * 4);   // BT*20 f32

    // K1: 65536 M-tiles / 8 per wave = 8192 waves / 8 per block = 1024 blocks.
    k1_xg_gemm<<<1024, 256, 0, stream>>>(x, W_ih, b_ih, b_hh, xg);

    // K2: scan. 256 sequences / 16 per wave -> 16 single-wave blocks.
    k2_scan<<<16, 32, 16384, stream>>>(xg, W_hh, rout);

    // K3: MLP head over BT rows.
    k3_mlp<<<(int)(BT / 256), 256, 0, stream>>>(rout, Wh, bh, W1, b1, W2, b2,
                                                W3, b3, out, BT);
}
